// SlotAttention_5617817224158
// MI455X (gfx1250) — matrix operand order repaired and checked by reading, compiled, not run on hardware
//
#include <hip/hip_runtime.h>

typedef __attribute__((ext_vector_type(16))) _Float16 v16h;
typedef __attribute__((ext_vector_type(8)))  float    v8f;

#define DIMD   512
#define NSLOT  8
#define NB     32
#define NAUD   1024
#define NIMG   3136
#define NITERS 3
#define EPSA   1e-8f
#define LN_EPS 1e-5f
#define DSCALE 0.044194173824159216f   /* 512^-0.5 */
#define LDSTH  520                     /* padded LDS row stride (halfs) */

__device__ __forceinline__ v8f wmma_f16(v16h a, v16h b, v8f c) {
  return __builtin_amdgcn_wmma_f32_16x16x32_f16(false, a, false, b, (short)0, c,
                                                false, false);
}

// ---- Fragment loaders (wave32 CDNA5 WMMA layouts) -------------------------
// A 16x32 (MxK): lane m = lane&15 ; K = h+grp*16
__device__ __forceinline__ v16h load_a_f32(const float* A, int lda, int m0,
                                           int kk, int mvalid) {
  int lane = threadIdx.x & 31;
  int grp  = lane >> 4;
  int mi   = lane & 15;
  v16h a;
  if (mi < mvalid) {
    const float* row = A + (size_t)(m0 + mi) * lda + kk;
#pragma unroll
    for (int h = 0; h < 16; h += 2) {
      int k    = h + grp * 16;
      a[h]     = (_Float16)row[k];
      a[h + 1] = (_Float16)row[k + 1];
    }
  } else {
#pragma unroll
    for (int h = 0; h < 16; ++h) a[h] = (_Float16)0.0f;
  }
  return a;
}

// A fragment from f16 LDS tile: one contiguous 16-half run -> 2x ds_load_b128
__device__ __forceinline__ v16h load_a_ldsh(const _Float16* lds, int kk) {
  int lane = threadIdx.x & 31;
  int grp  = lane >> 4;
  const _Float16* row = lds + (lane & 15) * LDSTH + kk + grp * 16;
  v16h a;
#pragma unroll
  for (int h = 0; h < 8; ++h) a[h] = row[h];          // K = kk+grp*16 .. +7
#pragma unroll
  for (int h = 8; h < 16; ++h) a[h] = row[h];         // K = kk+grp*16+8 .. +15
  return a;
}

// B 32x16 (KxN) where B[k][n] = W[n][k], W row-major f16 [N,K]:
// lane column n = lane&15, 16 contiguous K at kk+grp*16
__device__ __forceinline__ v16h load_b_wt(const _Float16* W, int ldk, int n0,
                                          int kk) {
  int lane = threadIdx.x & 31;
  const _Float16* p =
      W + (size_t)(n0 + (lane & 15)) * ldk + kk + (lane >> 4) * 16;
  v16h b;
#pragma unroll
  for (int h = 0; h < 16; ++h) b[h] = p[h];
  return b;
}

// B[k][n] = S[n][k], S row-major f32 (k-matrix for dots: tokens are columns)
__device__ __forceinline__ v16h load_b_tr_f32(const float* S, int ldk, int n0,
                                              int kk) {
  int lane = threadIdx.x & 31;
  const float* p = S + (size_t)(n0 + (lane & 15)) * ldk + kk + (lane >> 4) * 16;
  v16h b;
#pragma unroll
  for (int h = 0; h < 16; ++h) b[h] = (_Float16)p[h];
  return b;
}

// B[k][n] = V[k][n], V row-major f32 (v-matrix for attn@v: tokens are K)
__device__ __forceinline__ v16h load_b_f32(const float* V, int ldn, int n0,
                                           int kk) {
  int lane = threadIdx.x & 31;
  int n    = n0 + (lane & 15);
  int kb   = kk + (lane >> 4) * 16;
  v16h b;
#pragma unroll
  for (int h = 0; h < 16; ++h) b[h] = (_Float16)V[(size_t)(kb + h) * ldn + n];
  return b;
}

// ---- Kernels --------------------------------------------------------------

__global__ void sa_f16cvt(const float* __restrict__ x, _Float16* __restrict__ y,
                          int n) {
  int i = blockIdx.x * blockDim.x + threadIdx.x;
  if (i < n) y[i] = (_Float16)x[i];
}

__global__ void sa_init_slots(const float* __restrict__ noise,
                              const float* __restrict__ mu,
                              const float* __restrict__ logsig,
                              float* __restrict__ sa, float* __restrict__ si) {
  int idx = blockIdx.x * blockDim.x + threadIdx.x;  // 32*8*512
  int c   = idx & (DIMD - 1);
  float v = mu[c] + __expf(logsig[c]) * noise[idx];
  sa[idx] = v;
  si[idx] = v;
}

// One wave per row LayerNorm (256 rows)
__global__ __launch_bounds__(256) void sa_ln_rows(const float* __restrict__ X,
                                                  const float* __restrict__ g,
                                                  const float* __restrict__ bb,
                                                  float* __restrict__ Y) {
  int wave = threadIdx.x >> 5, lane = threadIdx.x & 31;
  int r    = blockIdx.x * 8 + wave;
  const float* x = X + (size_t)r * DIMD;
  float s = 0.f, s2 = 0.f, vals[16];
#pragma unroll
  for (int t = 0; t < 16; ++t) {
    float v = x[lane + 32 * t];
    vals[t] = v; s += v; s2 += v * v;
  }
#pragma unroll
  for (int off = 16; off; off >>= 1) {
    s  += __shfl_xor(s, off, 32);
    s2 += __shfl_xor(s2, off, 32);
  }
  float mean = s / DIMD;
  float rstd = rsqrtf(s2 / DIMD - mean * mean + LN_EPS);
#pragma unroll
  for (int t = 0; t < 16; ++t) {
    int c = lane + 32 * t;
    Y[(size_t)r * DIMD + c] = (vals[t] - mean) * rstd * g[c] + bb[c];
  }
}

// Fused LayerNorm + dual projection (k and v) for one 16-row tile.
// LN result stored as f16 in LDS (converted once); 8 waves, wave w covers
// output columns [w*64, w*64+64) of both k and v.
__global__ __launch_bounds__(256) void sa_ln_kv_proj(
    const float* __restrict__ X, const float* __restrict__ g,
    const float* __restrict__ bln, const _Float16* __restrict__ Wk16,
    const float* __restrict__ bk, const _Float16* __restrict__ Wv16,
    const float* __restrict__ bv, float* __restrict__ Kout,
    float* __restrict__ Vout) {
  __shared__ _Float16 ldsh[16 * LDSTH];
  int wave = threadIdx.x >> 5, lane = threadIdx.x & 31;
  int r0   = blockIdx.x * 16;
#pragma unroll
  for (int rr = 0; rr < 2; ++rr) {
    int r = wave * 2 + rr;
    const float* x = X + (size_t)(r0 + r) * DIMD;
    float s = 0.f, s2 = 0.f, vals[16];
#pragma unroll
    for (int t = 0; t < 16; ++t) {
      float v = x[lane + 32 * t];
      vals[t] = v; s += v; s2 += v * v;
    }
#pragma unroll
    for (int off = 16; off; off >>= 1) {
      s  += __shfl_xor(s, off, 32);
      s2 += __shfl_xor(s2, off, 32);
    }
    float mean = s / DIMD;
    float rstd = rsqrtf(s2 / DIMD - mean * mean + LN_EPS);
#pragma unroll
    for (int t = 0; t < 16; ++t) {
      int c = lane + 32 * t;
      ldsh[r * LDSTH + c] =
          (_Float16)((vals[t] - mean) * rstd * g[c] + bln[c]);
    }
  }
  __syncthreads();

  v8f z = {};
  v8f ak[4] = {z, z, z, z};
  v8f av[4] = {z, z, z, z};
  int nb = wave * 64;
  for (int kk = 0; kk < DIMD; kk += 32) {
    v16h a = load_a_ldsh(ldsh, kk);
    // all 8 B tiles in distinct registers so the global_load_b128 clause can
    // overlap with the WMMA drain instead of serializing on one reg pair
    v16h bks[4], bvs[4];
#pragma unroll
    for (int t = 0; t < 4; ++t) {
      bks[t] = load_b_wt(Wk16, DIMD, nb + t * 16, kk);
      bvs[t] = load_b_wt(Wv16, DIMD, nb + t * 16, kk);
    }
#pragma unroll
    for (int t = 0; t < 4; ++t) {
      ak[t] = wmma_f16(a, bks[t], ak[t]);
      av[t] = wmma_f16(a, bvs[t], av[t]);
    }
  }
  int grp = lane >> 4;
#pragma unroll
  for (int t = 0; t < 4; ++t) {
    int n = nb + t * 16 + (lane & 15);
#pragma unroll
    for (int r = 0; r < 8; ++r) {
      int m = r0 + r + grp * 8;
      Kout[(size_t)m * DIMD + n] = ak[t][r] + bk[n];
      Vout[(size_t)m * DIMD + n] = av[t][r] + bv[n];
    }
  }
}

// Generic small GEMM: C[M,N] = act(A[M,K] @ W16[N,K]^T + bias) (+ resid)
// K unrolled by 2 with distinct fragments for load/WMMA overlap.
__global__ __launch_bounds__(128) void sa_gemm(const float* __restrict__ A,
                                               const _Float16* __restrict__ W,
                                               const float* __restrict__ bias,
                                               const float* resid, float* C,
                                               int N, int K, int relu) {
  int wave = threadIdx.x >> 5, lane = threadIdx.x & 31;
  int m0   = blockIdx.y * 16;
  int n0   = blockIdx.x * 64 + wave * 16;
  v8f acc  = {};
  for (int kk = 0; kk < K; kk += 64) {
    v16h a0 = load_a_f32(A, K, m0, kk, 16);
    v16h b0 = load_b_wt(W, K, n0, kk);
    v16h a1 = load_a_f32(A, K, m0, kk + 32, 16);
    v16h b1 = load_b_wt(W, K, n0, kk + 32);
    acc = wmma_f16(a0, b0, acc);
    acc = wmma_f16(a1, b1, acc);
  }
  int n = n0 + (lane & 15);
  int grp = lane >> 4;
#pragma unroll
  for (int r = 0; r < 8; ++r) {
    int m   = m0 + r + grp * 8;
    float v = acc[r] + bias[n];
    if (relu) v = fmaxf(v, 0.0f);
    if (resid) v += resid[(size_t)m * N + n];
    C[(size_t)m * N + n] = v;
  }
}

// dots = (q @ k^T)*scale -> softmax over slot axis -> store a, atomic den
// One wave per (batch, 16-token tile). Slot rows 8..15 are zero-padded.
__global__ __launch_bounds__(32) void sa_attn_softmax(
    const float* __restrict__ Q, const float* __restrict__ Kmat,
    float* __restrict__ Aout, float* __restrict__ den, int N) {
  int b    = blockIdx.y;
  int j0   = blockIdx.x * 16;
  int lane = threadIdx.x & 31;
  const float* q = Q + (size_t)b * NSLOT * DIMD;
  const float* k = Kmat + (size_t)b * N * DIMD;
  v8f acc = {};
  for (int kk = 0; kk < DIMD; kk += 64) {
    v16h a0 = load_a_f32(q, DIMD, 0, kk, NSLOT);
    v16h b0 = load_b_tr_f32(k, DIMD, j0, kk);
    v16h a1 = load_a_f32(q, DIMD, 0, kk + 32, NSLOT);
    v16h b1 = load_b_tr_f32(k, DIMD, j0, kk + 32);
    acc = wmma_f16(a0, b0, acc);
    acc = wmma_f16(a1, b1, acc);
  }
  // lanes 0..15: token j0+lane, slot i lives in acc[i] (i=0..7)
  float e[NSLOT];
  float mx = -1e30f;
#pragma unroll
  for (int r = 0; r < NSLOT; ++r) {
    e[r] = acc[r] * DSCALE;
    mx   = fmaxf(mx, e[r]);
  }
  float s = 0.f;
#pragma unroll
  for (int r = 0; r < NSLOT; ++r) { e[r] = __expf(e[r] - mx); s += e[r]; }
  float inv   = 1.0f / s;
  int j       = j0 + (lane & 15);
  bool active = lane < 16;
#pragma unroll
  for (int r = 0; r < NSLOT; ++r) {
    float a = e[r] * inv;
    if (active) Aout[(size_t)b * NSLOT * N + (size_t)r * N + j] = a;
    float part = active ? a : 0.0f;
#pragma unroll
    for (int off = 8; off; off >>= 1) part += __shfl_xor(part, off, 32);
    if (lane == 0) atomicAdd(&den[b * NSLOT + r], part);
  }
}

// upd = (a @ v) / (den+eps). 4 waves cover 64 dims; K loop over all tokens.
__global__ __launch_bounds__(128) void sa_attn_av(const float* __restrict__ Am,
                                                  const float* __restrict__ V,
                                                  const float* __restrict__ den,
                                                  float* __restrict__ upd,
                                                  int N) {
  int b    = blockIdx.y;
  int wave = threadIdx.x >> 5, lane = threadIdx.x & 31;
  int n0   = blockIdx.x * 64 + wave * 16;
  const float* a_b = Am + (size_t)b * NSLOT * N;
  const float* v_b = V + (size_t)b * N * DIMD;
  v8f acc = {};
  for (int kk = 0; kk < N; kk += 32) {
    v16h a  = load_a_f32(a_b, N, 0, kk, NSLOT);
    v16h bb = load_b_f32(v_b, DIMD, n0, kk);
    acc     = wmma_f16(a, bb, acc);
  }
  if (lane < 16) {
    int n = n0 + lane;
#pragma unroll
    for (int r = 0; r < NSLOT; ++r) {
      float d = den[b * NSLOT + r] + EPSA;
      upd[(size_t)(b * NSLOT + r) * DIMD + n] = acc[r] / d;
    }
  }
}

__global__ void sa_attn_norm(float* A, const float* __restrict__ den, int N) {
  int row   = blockIdx.x;  // 0..255 = b*8+slot
  float inv = 1.0f / (den[row] + EPSA);
  for (int j = threadIdx.x; j < N; j += blockDim.x)
    A[(size_t)row * N + j] *= inv;
}

__global__ void sa_gru(const float* __restrict__ gi,
                       const float* __restrict__ gh, float* slots) {
  int idx = blockIdx.x * blockDim.x + threadIdx.x;  // 256*512
  int m = idx >> 9, c = idx & (DIMD - 1);
  const float* gim = gi + (size_t)m * 3 * DIMD;
  const float* ghm = gh + (size_t)m * 3 * DIMD;
  float r  = 1.f / (1.f + __expf(-(gim[c] + ghm[c])));
  float zz = 1.f / (1.f + __expf(-(gim[DIMD + c] + ghm[DIMD + c])));
  float n  = tanhf(gim[2 * DIMD + c] + r * ghm[2 * DIMD + c]);
  float h  = slots[idx];
  slots[idx] = (1.f - zz) * n + zz * h;
}

// ---- Host orchestration ---------------------------------------------------

struct StepBufs {
  float *sn, *upd, *hid, *gi, *gh, *den;
  const _Float16 *Wq16, *Wih16, *Whh16, *W1h, *W2h;
  const float *bq, *b_ih, *b_hh, *b1, *b2;
  const float *ln_slots_g, *ln_slots_b, *ln_ff_g, *ln_ff_b;
};

static void run_step(hipStream_t stream, float* slots, int N, const float* kmat,
                     const float* vmat, float* q_out, float* attn_out,
                     const StepBufs& s) {
  sa_ln_rows<<<32, 256, 0, stream>>>(slots, s.ln_slots_g, s.ln_slots_b, s.sn);
  sa_gemm<<<dim3(8, 16), 128, 0, stream>>>(s.sn, s.Wq16, s.bq, nullptr, q_out,
                                           512, 512, 0);
  hipMemsetAsync(s.den, 0, NB * NSLOT * sizeof(float), stream);
  sa_attn_softmax<<<dim3(N / 16, NB), 32, 0, stream>>>(q_out, kmat, attn_out,
                                                       s.den, N);
  sa_attn_av<<<dim3(8, NB), 128, 0, stream>>>(attn_out, vmat, s.den, s.upd, N);
  sa_attn_norm<<<NB * NSLOT, 256, 0, stream>>>(attn_out, s.den, N);
  sa_gemm<<<dim3(24, 16), 128, 0, stream>>>(s.upd, s.Wih16, s.b_ih, nullptr,
                                            s.gi, 1536, 512, 0);
  sa_gemm<<<dim3(24, 16), 128, 0, stream>>>(slots, s.Whh16, s.b_hh, nullptr,
                                            s.gh, 1536, 512, 0);
  sa_gru<<<512, 256, 0, stream>>>(s.gi, s.gh, slots);
  sa_ln_rows<<<32, 256, 0, stream>>>(slots, s.ln_ff_g, s.ln_ff_b, s.sn);
  sa_gemm<<<dim3(8, 16), 128, 0, stream>>>(s.sn, s.W1h, s.b1, nullptr, s.hid,
                                           512, 512, 1);
  sa_gemm<<<dim3(8, 16), 128, 0, stream>>>(s.hid, s.W2h, s.b2, slots, slots,
                                           512, 512, 0);
}

extern "C" void kernel_launch(void* const* d_in, const int* in_sizes, int n_in,
                              void* d_out, int out_size, void* d_ws,
                              size_t ws_size, hipStream_t stream) {
  (void)in_sizes; (void)n_in; (void)out_size; (void)ws_size;
  const float* in_audio  = (const float*)d_in[0];
  const float* in_image  = (const float*)d_in[1];
  const float* noise     = (const float*)d_in[2];
  const float* mu        = (const float*)d_in[3];
  const float* logsig    = (const float*)d_in[4];
  const float* ln_in_g   = (const float*)d_in[5];
  const float* ln_in_b   = (const float*)d_in[6];
  const float* ln_sl_g   = (const float*)d_in[7];
  const float* ln_sl_b   = (const float*)d_in[8];
  const float* ln_ff_g   = (const float*)d_in[9];
  const float* ln_ff_b   = (const float*)d_in[10];
  const float* Wq        = (const float*)d_in[11];
  const float* bq        = (const float*)d_in[12];
  const float* Wk        = (const float*)d_in[13];
  const float* bk        = (const float*)d_in[14];
  const float* Wv        = (const float*)d_in[15];
  const float* bv        = (const float*)d_in[16];
  const float* W_ih      = (const float*)d_in[17];
  const float* W_hh      = (const float*)d_in[18];
  const float* b_ih      = (const float*)d_in[19];
  const float* b_hh      = (const float*)d_in[20];
  const float* W1        = (const float*)d_in[21];
  const float* b1        = (const float*)d_in[22];
  const float* W2        = (const float*)d_in[23];
  const float* b2        = (const float*)d_in[24];

  float* out        = (float*)d_out;
  float* o_slots_a  = out;                    // 131072
  float* o_q_a      = out + 131072;           // 131072
  float* o_k_a      = out + 262144;           // 16777216
  float* o_attn_a   = out + 17039360;         // 262144
  float* o_cross_ia = out + 17301504;         // 262144
  float* o_slots_i  = out + 17563648;         // 131072
  float* o_q_i      = out + 17694720;         // 131072
  float* o_k_i      = out + 17825792;         // 51380224
  float* o_attn_i   = out + 69206016;         // 802816
  float* o_cross_ai = out + 70008832;         // 802816

  // workspace carve-up
  _Float16* Wq16  = (_Float16*)d_ws;
  _Float16* Wk16  = Wq16 + 262144;
  _Float16* Wv16  = Wk16 + 262144;
  _Float16* Wih16 = Wv16 + 262144;
  _Float16* Whh16 = Wih16 + 786432;
  _Float16* W1h   = Whh16 + 786432;
  _Float16* W2h   = W1h + 262144;
  float* v_a = (float*)(W2h + 262144);
  float* v_i = v_a + 16777216;
  float* sn  = v_i + 51380224;
  float* upd = sn + 131072;
  float* hid = upd + 131072;
  float* gi  = hid + 131072;
  float* gh  = gi + 393216;
  float* den = gh + 393216;

  // weight conversions to f16 (one-time per call)
  sa_f16cvt<<<1024, 256, 0, stream>>>(Wq, Wq16, 262144);
  sa_f16cvt<<<1024, 256, 0, stream>>>(Wk, Wk16, 262144);
  sa_f16cvt<<<1024, 256, 0, stream>>>(Wv, Wv16, 262144);
  sa_f16cvt<<<3072, 256, 0, stream>>>(W_ih, Wih16, 786432);
  sa_f16cvt<<<3072, 256, 0, stream>>>(W_hh, Whh16, 786432);
  sa_f16cvt<<<1024, 256, 0, stream>>>(W1, W1h, 262144);
  sa_f16cvt<<<1024, 256, 0, stream>>>(W2, W2h, 262144);

  sa_init_slots<<<512, 256, 0, stream>>>(noise, mu, logsig, o_slots_a,
                                         o_slots_i);

  // fused LN + K/V projections (the FLOP/byte-dominant stage)
  sa_ln_kv_proj<<<(NB * NAUD) / 16, 256, 0, stream>>>(
      in_audio, ln_in_g, ln_in_b, Wk16, bk, Wv16, bv, o_k_a, v_a);
  sa_ln_kv_proj<<<(NB * NIMG) / 16, 256, 0, stream>>>(
      in_image, ln_in_g, ln_in_b, Wk16, bk, Wv16, bv, o_k_i, v_i);

  StepBufs s;
  s.sn = sn; s.upd = upd; s.hid = hid; s.gi = gi; s.gh = gh; s.den = den;
  s.Wq16 = Wq16; s.Wih16 = Wih16; s.Whh16 = Whh16; s.W1h = W1h; s.W2h = W2h;
  s.bq = bq; s.b_ih = b_ih; s.b_hh = b_hh; s.b1 = b1; s.b2 = b2;
  s.ln_slots_g = ln_sl_g; s.ln_slots_b = ln_sl_b;
  s.ln_ff_g = ln_ff_g; s.ln_ff_b = ln_ff_b;

  for (int it = 0; it < NITERS; ++it) {
    run_step(stream, o_slots_a, NAUD, o_k_a, v_a, o_q_a, o_attn_a, s);
    run_step(stream, o_slots_i, NIMG, o_k_i, v_i, o_q_i, o_attn_i, s);
  }

  // cross attentions: cross_ai = q_a -> k_i (N=3136), cross_ia = q_i -> k_a
  hipMemsetAsync(den, 0, NB * NSLOT * sizeof(float), stream);
  sa_attn_softmax<<<dim3(NIMG / 16, NB), 32, 0, stream>>>(o_q_a, o_k_i,
                                                          o_cross_ai, den, NIMG);
  sa_attn_norm<<<NB * NSLOT, 256, 0, stream>>>(o_cross_ai, den, NIMG);
  hipMemsetAsync(den, 0, NB * NSLOT * sizeof(float), stream);
  sa_attn_softmax<<<dim3(NAUD / 16, NB), 32, 0, stream>>>(o_q_i, o_k_a,
                                                          o_cross_ia, den, NAUD);
  sa_attn_norm<<<NB * NSLOT, 256, 0, stream>>>(o_cross_ia, den, NAUD);
}
